// GraphEvalGT_12979391169037
// MI455X (gfx1250) — matrix operand order repaired and checked by reading, not compiled, not run on hardware
//
#include <hip/hip_runtime.h>
#include <hip/hip_bf16.h>
#include <math.h>

typedef __attribute__((ext_vector_type(16))) __bf16 bf16x16;
typedef __attribute__((ext_vector_type(8)))  __bf16 bf16x8;
typedef __attribute__((ext_vector_type(8)))  float  f32x8;

#define GN 8192
#define GE 262144
#define GIN 64
#define GH 128
#define GOUT 10
#define GL 4
#define GMAXD 11
#define GROLES 25
#define GNG 32

// ---------------- elementwise utility kernels ----------------

__global__ void k_convert_f32_bf16(const float* __restrict__ in, __bf16* __restrict__ out, int n) {
  int i = blockIdx.x * blockDim.x + threadIdx.x;
  if (i < n) out[i] = (__bf16)in[i];
}

__global__ void k_fill_f32(float* __restrict__ p, float v, int n) {
  int i = blockIdx.x * blockDim.x + threadIdx.x;
  if (i < n) p[i] = v;
}

__global__ void k_add_inplace(float* __restrict__ a, const float* __restrict__ b, int n) {
  int i = blockIdx.x * blockDim.x + threadIdx.x;
  if (i < n) a[i] += b[i];
}

// ---------------- weight pack: f32 [K x Nc] -> bf16 WMMA B-fragment order ----------------
// Padded to Ncp = ceil(Nc/64)*64 columns (zeros). Layout: [ktile][ntile][lane][j] where
// lane = (kin/16)*16 + (n%16), j = kin%16  (contiguous k order, same order as the A fragment loads).
__global__ void k_pack_b(const float* __restrict__ W, __bf16* __restrict__ P,
                         int K, int Nc, int Ncp) {
  int idx = blockIdx.x * blockDim.x + threadIdx.x;
  if (idx >= K * Ncp) return;
  int k = idx / Ncp;
  int n = idx - k * Ncp;
  float val = (n < Nc) ? W[(size_t)k * Nc + n] : 0.0f;
  int ktile = k >> 5, kin = k & 31;
  int ntile = n >> 4, nin = n & 15;
  int lane = ((kin >> 4) << 4) | nin;
  int j = kin & 15;
  size_t dst = (((size_t)ktile * (Ncp >> 4) + ntile) * 32 + lane) * 16 + j;
  P[dst] = (__bf16)val;
}

// ---------------- bf16 WMMA GEMM: C = act(A*Bp + bias) (+ residual) ----------------
// A: [M x K] row-major bf16. Bp: packed fragments, Ncp columns (mult of 64).
// Block = 4 waves; all waves share one 64-column panel of B, staged in LDS once.
// Each wave computes a 16x64 tile (4 accumulators, A fragment reused 4x, B from LDS).
__global__ void __launch_bounds__(128)
k_gemm_bf16_wmma(const __bf16* __restrict__ A, const __bf16* __restrict__ Bp,
                 const float* __restrict__ bias, const float* __restrict__ residual,
                 float* __restrict__ outF, __bf16* __restrict__ outBF,
                 int M, int K, int Nc, int Ncp, int act) {
  __shared__ __bf16 sB[8 * 2048];      // up to K=256 (8 k-tiles * 4KB) = 32KB
  int tid  = threadIdx.x;
  int wave = tid >> 5;
  int lane = tid & 31;
  int n0   = blockIdx.x * 64;
  int ntiles = Ncp >> 4;
  int ktiles = K >> 5;

  // cooperative stage: for each k-tile the 4 B tiles are 2048 contiguous bf16
  int chunks = ktiles << 8;            // 16B chunks (8 bf16) per panel
  for (int c = tid; c < chunks; c += 128) {
    int ktile = c >> 8;
    int i = (c & 255) << 3;
    *(bf16x8*)(sB + ktile * 2048 + i) =
        *(const bf16x8*)(Bp + ((size_t)ktile * ntiles + (n0 >> 4)) * 512 + i);
  }
  __syncthreads();

  int m0 = blockIdx.y * 64 + wave * 16;
  if (m0 >= M) return;                 // wave-uniform exit (after barrier)

  int half = lane >> 4;                // 0: lanes 0-15, 1: lanes 16-31
  int mn   = lane & 15;                // A row / C column within tile
  const __bf16* Arow = A + (size_t)(m0 + mn) * K;

  f32x8 c0 = {0.f,0.f,0.f,0.f,0.f,0.f,0.f,0.f};
  f32x8 c1 = c0, c2 = c0, c3 = c0;

  for (int k0 = 0; k0 < K; k0 += 32) {
    // A fragment: half 0 -> K {0..15}, half 1 -> K {16..31}
    bf16x8 a0 = *(const bf16x8*)(Arow + k0 + half * 16);
    bf16x8 a1 = *(const bf16x8*)(Arow + k0 + half * 16 + 8);
    bf16x16 a;
#pragma unroll
    for (int j = 0; j < 8; ++j) { a[j] = a0[j]; a[8 + j] = a1[j]; }

    // 4 B fragments from LDS: contiguous 32B per lane per tile
    const __bf16* Bt = sB + (k0 >> 5) * 2048 + lane * 16;
    bf16x16 b0 = *(const bf16x16*)(Bt);
    bf16x16 b1 = *(const bf16x16*)(Bt + 512);
    bf16x16 b2 = *(const bf16x16*)(Bt + 1024);
    bf16x16 b3 = *(const bf16x16*)(Bt + 1536);

    c0 = __builtin_amdgcn_wmma_f32_16x16x32_bf16(false, a, false, b0, (short)0, c0, false, false);
    c1 = __builtin_amdgcn_wmma_f32_16x16x32_bf16(false, a, false, b1, (short)0, c1, false, false);
    c2 = __builtin_amdgcn_wmma_f32_16x16x32_bf16(false, a, false, b2, (short)0, c2, false, false);
    c3 = __builtin_amdgcn_wmma_f32_16x16x32_bf16(false, a, false, b3, (short)0, c3, false, false);
  }

  f32x8 acc[4] = {c0, c1, c2, c3};
#pragma unroll
  for (int t = 0; t < 4; ++t) {
    int col = n0 + t * 16 + mn;
    bool cok = col < Nc;
    float bv = (bias && cok) ? bias[col] : 0.0f;
#pragma unroll
    for (int r = 0; r < 8; ++r) {
      int row = m0 + half * 8 + r;     // C/D layout: VGPR r -> M = r + 8*half
      if (row < M && cok) {
        float v = acc[t][r] + bv;
        if (act == 1) v = 0.5f * v * (1.0f + erff(v * 0.70710678118654752f)); // exact GELU
        else if (act == 2) v = fmaxf(v, 0.0f);
        if (residual) v += residual[(size_t)row * Nc + col];
        size_t oi = (size_t)row * Nc + col;
        if (outF)  outF[oi]  = v;
        if (outBF) outBF[oi] = (__bf16)v;
      }
    }
  }
}

// ---------------- LayerNorm (wave per row, H=128 -> 4 f32 per lane) ----------------
__global__ void __launch_bounds__(256)
k_ln_to_bf16(const float* __restrict__ h, const float* __restrict__ s,
             const float* __restrict__ b, __bf16* __restrict__ out, int n) {
  int wid  = (blockIdx.x * blockDim.x + threadIdx.x) >> 5;
  int lane = threadIdx.x & 31;
  if (wid >= n) return;
  float4 v = ((const float4*)(h + (size_t)wid * GH))[lane];
  float sum = v.x + v.y + v.z + v.w;
  float sq  = v.x * v.x + v.y * v.y + v.z * v.z + v.w * v.w;
#pragma unroll
  for (int off = 16; off > 0; off >>= 1) {
    sum += __shfl_xor(sum, off);
    sq  += __shfl_xor(sq,  off);
  }
  float mean = sum * (1.0f / GH);
  float var  = sq * (1.0f / GH) - mean * mean;
  float inv  = rsqrtf(var + 1e-5f);
  int base = lane * 4;
  size_t o = (size_t)wid * GH + base;
  out[o + 0] = (__bf16)((v.x - mean) * inv * s[base + 0] + b[base + 0]);
  out[o + 1] = (__bf16)((v.y - mean) * inv * s[base + 1] + b[base + 1]);
  out[o + 2] = (__bf16)((v.z - mean) * inv * s[base + 2] + b[base + 2]);
  out[o + 3] = (__bf16)((v.w - mean) * inv * s[base + 3] + b[base + 3]);
}

// ---------------- edge phase ----------------
__device__ __forceinline__ unsigned f2key(float f) {
  unsigned bits = __float_as_uint(f);
  return (bits & 0x80000000u) ? ~bits : (bits | 0x80000000u);
}
__device__ __forceinline__ float key2f(unsigned key) {
  unsigned bits = (key & 0x80000000u) ? (key & 0x7FFFFFFFu) : ~key;
  return __uint_as_float(bits);
}

// wave per edge: score = scale*dot(q[dst],k[src]) + fp[p] + fd[d]; atomicMax into mkey[dst]
__global__ void __launch_bounds__(256)
k_edge_score(const float* __restrict__ q, const float* __restrict__ k,
             const int* __restrict__ src, const int* __restrict__ dst,
             const int* __restrict__ pmat, const int* __restrict__ dmat,
             const float* __restrict__ fp, const float* __restrict__ fd,
             float* __restrict__ score, unsigned* __restrict__ mkey) {
  int e    = (blockIdx.x * blockDim.x + threadIdx.x) >> 5;
  int lane = threadIdx.x & 31;
  if (e >= GE) return;
  int s = src[e], d = dst[e];
  float4 qv = ((const float4*)(q + (size_t)d * GH))[lane];
  float4 kv = ((const float4*)(k + (size_t)s * GH))[lane];
  float p = qv.x * kv.x + qv.y * kv.y + qv.z * kv.z + qv.w * kv.w;
#pragma unroll
  for (int off = 16; off > 0; off >>= 1) p += __shfl_xor(p, off);
  if (lane == 0) {
    long long pi = (long long)s * GN + d;
    float sc = p * 0.08838834764831845f + fp[pmat[pi]] + fd[dmat[pi]];
    score[e] = sc;
    atomicMax(&mkey[d], f2key(sc));
  }
}

// thread per edge: e = exp(score - max[dst]); denom[dst] += e
__global__ void k_edge_exp(float* __restrict__ score, const int* __restrict__ dst,
                           const unsigned* __restrict__ mkey, float* __restrict__ denom) {
  int e = blockIdx.x * blockDim.x + threadIdx.x;
  if (e >= GE) return;
  int d = dst[e];
  float ex = expf(score[e] - key2f(mkey[d]));
  score[e] = ex;
  atomicAdd(&denom[d], ex);
}

// wave per edge: agg[dst] += alpha * v[src]
__global__ void __launch_bounds__(256)
k_edge_agg(const float* __restrict__ v, const float* __restrict__ escore,
           const int* __restrict__ src, const int* __restrict__ dst,
           const float* __restrict__ denom, float* __restrict__ agg) {
  int e    = (blockIdx.x * blockDim.x + threadIdx.x) >> 5;
  int lane = threadIdx.x & 31;
  if (e >= GE) return;
  int s = src[e], d = dst[e];
  float alpha = escore[e] / (denom[d] + 1e-16f);
  float4 vv = ((const float4*)(v + (size_t)s * GH))[lane];
  float* ap = agg + (size_t)d * GH + lane * 4;
  atomicAdd(ap + 0, vv.x * alpha);
  atomicAdd(ap + 1, vv.y * alpha);
  atomicAdd(ap + 2, vv.z * alpha);
  atomicAdd(ap + 3, vv.w * alpha);
}

// ---------------- global mean pool ----------------
__global__ void k_pool_accum(const float* __restrict__ h, const int* __restrict__ batch,
                             float* __restrict__ sums, float* __restrict__ cnt) {
  int idx = blockIdx.x * blockDim.x + threadIdx.x;
  if (idx >= GN * GH) return;
  int n = idx >> 7, i = idx & 127;
  int g = batch[n];
  atomicAdd(&sums[(size_t)g * GH + i], h[idx]);
  if (i == 0) atomicAdd(&cnt[g], 1.0f);
}

__global__ void k_pool_final(const float* __restrict__ sums, const float* __restrict__ cnt,
                             __bf16* __restrict__ g_bf16) {
  int idx = blockIdx.x * blockDim.x + threadIdx.x;
  if (idx >= GNG * GH) return;
  g_bf16[idx] = (__bf16)(sums[idx] / fmaxf(cnt[idx >> 7], 1.0f));
}

// ---------------- launcher ----------------
static inline int padN(int n) { return (n + 63) & ~63; }

extern "C" void kernel_launch(void* const* d_in, const int* in_sizes, int n_in,
                              void* d_out, int out_size, void* d_ws, size_t ws_size,
                              hipStream_t stream) {
  const float* x     = (const float*)d_in[0];
  const int*   ei    = (const int*)d_in[1];
  const int*   src   = ei;
  const int*   dstI  = ei + GE;
  const int*   pmat  = (const int*)d_in[2];
  const int*   dmat  = (const int*)d_in[3];
  const int*   batch = (const int*)d_in[4];
  const float* x_W   = (const float*)d_in[5];
  const float* in_b  = (const float*)d_in[6];
  const float* ln1_s = (const float*)d_in[7];
  const float* ln1_b = (const float*)d_in[8];
  const float* Wq    = (const float*)d_in[9];
  const float* Wk    = (const float*)d_in[10];
  const float* Wv    = (const float*)d_in[11];
  const float* fp    = (const float*)d_in[12];
  const float* fd    = (const float*)d_in[13];
  const float* ln2_s = (const float*)d_in[14];
  const float* ln2_b = (const float*)d_in[15];
  const float* W1    = (const float*)d_in[16];
  const float* b1    = (const float*)d_in[17];
  const float* W2    = (const float*)d_in[18];
  const float* b2    = (const float*)d_in[19];
  const float* cW1   = (const float*)d_in[20];
  const float* cb1   = (const float*)d_in[21];
  const float* cW2   = (const float*)d_in[22];
  const float* cb2   = (const float*)d_in[23];

  char* ws = (char*)d_ws;
  size_t off = 0;
  auto alloc = [&](size_t bytes) -> char* {
    char* p = ws + off;
    off += (bytes + 255) & ~(size_t)255;
    return p;
  };
  float*    h     = (float*)alloc((size_t)GN * GH * 4);
  float*    q     = (float*)alloc((size_t)GN * GH * 4);
  float*    k     = (float*)alloc((size_t)GN * GH * 4);
  float*    v     = (float*)alloc((size_t)GN * GH * 4);
  float*    agg   = (float*)alloc((size_t)GN * GH * 4);
  float*    score = (float*)alloc((size_t)GE * 4);
  unsigned* mkey  = (unsigned*)alloc((size_t)GN * 4);
  float*    denom = (float*)alloc((size_t)GN * 4);
  __bf16*   hn    = (__bf16*)alloc((size_t)GN * GH * 2);
  __bf16*   mid   = (__bf16*)alloc((size_t)GN * 2 * GH * 2);
  __bf16*   xb    = (__bf16*)alloc((size_t)GN * GIN * 2);
  __bf16*   pinW  = (__bf16*)alloc((size_t)GIN * padN(GH) * 2);
  __bf16*   pWq   = (__bf16*)alloc((size_t)GL * GH * padN(GH) * 2);
  __bf16*   pWk   = (__bf16*)alloc((size_t)GL * GH * padN(GH) * 2);
  __bf16*   pWv   = (__bf16*)alloc((size_t)GL * GH * padN(GH) * 2);
  __bf16*   pW1   = (__bf16*)alloc((size_t)GL * GH * padN(2 * GH) * 2);
  __bf16*   pW2   = (__bf16*)alloc((size_t)GL * 2 * GH * padN(GH) * 2);
  __bf16*   pc1   = (__bf16*)alloc((size_t)GH * padN(GH / 2) * 2);
  __bf16*   pc2   = (__bf16*)alloc((size_t)(GH / 2) * padN(GOUT) * 2);
  float*    sums  = (float*)alloc((size_t)GNG * GH * 4);
  float*    cnt   = (float*)alloc((size_t)GNG * 4);
  __bf16*   gb    = (__bf16*)alloc((size_t)GNG * GH * 2);
  __bf16*   t1    = (__bf16*)alloc((size_t)GNG * (GH / 2) * 2);

  auto fill = [&](float* p, float val, int n) {
    k_fill_f32<<<(n + 255) / 256, 256, 0, stream>>>(p, val, n);
  };
  auto pack = [&](const float* W, __bf16* P, int K, int Nc) {
    int n = K * padN(Nc);
    k_pack_b<<<(n + 255) / 256, 256, 0, stream>>>(W, P, K, Nc, padN(Nc));
  };
  auto gemm = [&](const __bf16* A, const __bf16* Bpk, const float* bias, const float* res,
                  float* oF, __bf16* oBF, int M, int K, int Nc, int act) {
    int Ncp = padN(Nc);
    dim3 g(Ncp / 64, (M + 63) / 64);
    k_gemm_bf16_wmma<<<g, 128, 0, stream>>>(A, Bpk, bias, res, oF, oBF, M, K, Nc, Ncp, act);
  };

  // input conversion + weight packing (recomputed every call -> deterministic, capture-safe)
  k_convert_f32_bf16<<<(GN * GIN + 255) / 256, 256, 0, stream>>>(x, xb, GN * GIN);
  pack(x_W, pinW, GIN, GH);
  for (int l = 0; l < GL; ++l) {
    pack(Wq + (size_t)l * GH * GH, pWq + (size_t)l * GH * padN(GH), GH, GH);
    pack(Wk + (size_t)l * GH * GH, pWk + (size_t)l * GH * padN(GH), GH, GH);
    pack(Wv + (size_t)l * GH * GH, pWv + (size_t)l * GH * padN(GH), GH, GH);
    pack(W1 + (size_t)l * GH * 2 * GH, pW1 + (size_t)l * GH * padN(2 * GH), GH, 2 * GH);
    pack(W2 + (size_t)l * 2 * GH * GH, pW2 + (size_t)l * 2 * GH * padN(GH), 2 * GH, GH);
  }
  pack(cW1, pc1, GH, GH / 2);
  pack(cW2, pc2, GH / 2, GOUT);

  // input projection: h = x @ in_W + in_b
  gemm(xb, pinW, in_b, nullptr, h, nullptr, GN, GIN, GH, 0);

  const int edgeWaveBlocks = (GE * 32 + 255) / 256;  // wave per edge, 8 edges/block
  for (int l = 0; l < GL; ++l) {
    // LN1 -> bf16
    k_ln_to_bf16<<<GN / 8, 256, 0, stream>>>(h, ln1_s + l * GH, ln1_b + l * GH, hn, GN);
    // QKV
    gemm(hn, pWq + (size_t)l * GH * padN(GH), nullptr, nullptr, q, nullptr, GN, GH, GH, 0);
    gemm(hn, pWk + (size_t)l * GH * padN(GH), nullptr, nullptr, k, nullptr, GN, GH, GH, 0);
    gemm(hn, pWv + (size_t)l * GH * padN(GH), nullptr, nullptr, v, nullptr, GN, GH, GH, 0);
    // reset segment buffers
    fill((float*)mkey, 0.0f, GN);           // key 0 == below every real score
    fill(denom, 0.0f, GN);
    fill(agg, 0.0f, GN * GH);
    // edge attention
    k_edge_score<<<edgeWaveBlocks, 256, 0, stream>>>(q, k, src, dstI, pmat, dmat,
                                                     fp + l * GMAXD, fd + l * GROLES,
                                                     score, mkey);
    k_edge_exp<<<(GE + 255) / 256, 256, 0, stream>>>(score, dstI, mkey, denom);
    k_edge_agg<<<edgeWaveBlocks, 256, 0, stream>>>(v, score, src, dstI, denom, agg);
    // residual
    k_add_inplace<<<(GN * GH + 255) / 256, 256, 0, stream>>>(h, agg, GN * GH);
    // FFN: mid = GELU(LN2(h) @ W1 + b1) [bf16]; h = h + mid @ W2 + b2
    k_ln_to_bf16<<<GN / 8, 256, 0, stream>>>(h, ln2_s + l * GH, ln2_b + l * GH, hn, GN);
    gemm(hn, pW1 + (size_t)l * GH * padN(2 * GH), b1 + l * 2 * GH, nullptr,
         nullptr, mid, GN, GH, 2 * GH, 1);
    gemm(mid, pW2 + (size_t)l * 2 * GH * padN(GH), b2 + l * GH, h,
         h, nullptr, GN, 2 * GH, GH, 0);
  }

  // global mean pool + classifier
  fill(sums, 0.0f, GNG * GH);
  fill(cnt, 0.0f, GNG);
  k_pool_accum<<<(GN * GH + 255) / 256, 256, 0, stream>>>(h, batch, sums, cnt);
  k_pool_final<<<(GNG * GH + 255) / 256, 256, 0, stream>>>(sums, cnt, gb);
  gemm(gb, pc1, cb1, nullptr, nullptr, t1, GNG, GH, GH / 2, 2);         // ReLU
  gemm(t1, pc2, cb2, nullptr, (float*)d_out, nullptr, GNG, GH / 2, GOUT, 0);
  (void)in_sizes; (void)n_in; (void)out_size; (void)ws_size;
}
